// DeformConv_5669356832967
// MI455X (gfx1250) — matrix operand order repaired and checked by reading, not compiled, not run on hardware
//
#include <hip/hip_runtime.h>
#include <hip/hip_fp16.h>

typedef __attribute__((ext_vector_type(16))) _Float16 v16h;
typedef __attribute__((ext_vector_type(8)))  float    v8f;

#define Bn   8
#define Cn   64
#define Hn   96
#define Wn   96
#define COUT 64
#define HW   (Hn * Wn)                  // 9216
#define NTILES (Bn * Hn * (Wn / 16))    // 4608
#define AW_BYTES (18 * 4 * 32 * 16 * 2) // 73728 B staged weights (f16)

// ---------------------------------------------------------------------------
// Kernel 1: stage weights f32 -> f16 in the exact per-lane WMMA A-fragment
// layout. aw[chunk][mt][lane] = v16h fragment for that lane.
// chunk = k2*2 + half (half selects channels 0..31 / 32..63).
// 16-bit A 16x32, contiguous k order chosen here (same as B): lanes 0-15 -> K{0..15}, lanes 16-31 -> K{16..31}
// ---------------------------------------------------------------------------
__global__ void stage_weights_kernel(const float* __restrict__ w,
                                     _Float16* __restrict__ aw) {
    int slot = blockIdx.x * blockDim.x + threadIdx.x;   // 0..2303
    if (slot >= 18 * 4 * 32) return;
    int lane  = slot & 31;
    int mt    = (slot >> 5) & 3;
    int chunk = slot >> 7;
    int k2    = chunk >> 1;
    int c0    = (chunk & 1) * 32;
    int mrow  = mt * 16 + (lane & 15);
    v16h frag;
#pragma unroll
    for (int j = 0; j < 16; ++j) {
        int kk = (lane < 16) ? j : j + 16;
        frag[j] = (_Float16)w[(mrow * Cn + c0 + kk) * 9 + k2];
    }
    *((v16h*)(aw) + slot) = frag;
}

// ---------------------------------------------------------------------------
// Kernel 2: NCHW -> NHWC transpose of x via padded LDS tile (conflict-free).
// One block per (b,h) row: 64 channels x 96 columns.
// ---------------------------------------------------------------------------
__global__ void nchw_to_nhwc_kernel(const float* __restrict__ x,
                                    float* __restrict__ xt) {
    __shared__ float tile[Wn * 65];     // 24960 B, pad 65 to rotate banks
    int bh = blockIdx.x;
    int b = bh / Hn, h = bh % Hn;
    const float* src = x + (size_t)b * Cn * HW + (size_t)h * Wn;
    float* dst = xt + (size_t)bh * Wn * Cn;

    for (int idx = threadIdx.x; idx < Cn * Wn; idx += blockDim.x) {
        int c = idx / Wn, w = idx - c * Wn;  // consecutive tid -> consecutive w (coalesced)
        tile[w * 65 + c] = src[(size_t)c * HW + w];
    }
    __syncthreads();
    for (int idx = threadIdx.x; idx < Cn * Wn; idx += blockDim.x) {
        int w = idx >> 6, c = idx & 63;      // consecutive tid -> consecutive c (coalesced)
        dst[idx] = tile[w * 65 + c];
    }
}

// ---------------------------------------------------------------------------
// Kernel 3: deformable conv via WMMA. One wave per 16-pixel tile, all 64
// output channels accumulated in 4 x v8f. Weight fragments served from LDS
// (dynamic, 73728 B) so the VMEM path is dedicated to bilinear gathers.
// ---------------------------------------------------------------------------
__global__ void __launch_bounds__(256)
deform_conv_wmma_kernel(const float* __restrict__ xt,
                        const float* __restrict__ offset,
                        const _Float16* __restrict__ aw,
                        float* __restrict__ out) {
    extern __shared__ _Float16 aLds[];   // 73728 B = all staged A fragments

    // cooperative copy of staged weights into LDS (b128 in, ds_store_b128 out)
    {
        const uint4* src = (const uint4*)aw;
        uint4* dst = (uint4*)aLds;
#pragma unroll
        for (int i = 0; i < AW_BYTES / 16 / 256; ++i)
            dst[i * 256 + threadIdx.x] = src[i * 256 + threadIdx.x];
    }
    __syncthreads();

    const int lane   = threadIdx.x & 31;
    const int tile   = blockIdx.x * (blockDim.x >> 5) + (threadIdx.x >> 5);
    const int wt     = tile % 6;
    const int ho     = (tile / 6) % Hn;
    const int b      = tile / (6 * Hn);
    const int wo     = wt * 16 + (lane & 15);   // pixel column for this lane (N index)
    const int laneHi = lane >> 4;               // 0: K 0..15 / M row,  1: K 16..31 / M row+8

    v8f acc[4];
#pragma unroll
    for (int mt = 0; mt < 4; ++mt) acc[mt] = (v8f){0.f,0.f,0.f,0.f,0.f,0.f,0.f,0.f};

    const float* offBase = offset + (((size_t)b * 18) * Hn + ho) * Wn + wo;
    const float* xtb     = xt + (size_t)b * HW * Cn;

#pragma unroll 1
    for (int k2 = 0; k2 < 9; ++k2) {
        const int kh = k2 / 3, kw = k2 - kh * 3;
        // offsets for this tap at this pixel (y channel = 2*k2, x channel = 2*k2+1)
        const float offy = __builtin_nontemporal_load(offBase + (size_t)(2 * k2)     * HW);
        const float offx = __builtin_nontemporal_load(offBase + (size_t)(2 * k2 + 1) * HW);
        // prefetch next tap's offset rows (global_prefetch_b8); clamp to stay in-bounds
        const int pk = (2 * k2 + 2 < 17) ? (2 * k2 + 2) : 16;
        __builtin_prefetch(offBase + (size_t)pk * HW, 0, 1);

        const float py = (float)(kh - 1 + ho) + offy;
        const float px = (float)(kw - 1 + wo) + offx;
        const float y0f = floorf(py), x0f = floorf(px);
        const float ly = py - y0f,  lx = px - x0f;
        const int y0 = (int)y0f, x0 = (int)x0f;
        const int y1 = y0 + 1,   x1 = x0 + 1;
        const float my0 = (y0 >= 0 && y0 < Hn) ? 1.f : 0.f;
        const float my1 = (y1 >= 0 && y1 < Hn) ? 1.f : 0.f;
        const float mx0 = (x0 >= 0 && x0 < Wn) ? 1.f : 0.f;
        const float mx1 = (x1 >= 0 && x1 < Wn) ? 1.f : 0.f;
        const float w00 = (1.f - ly) * (1.f - lx) * my0 * mx0;
        const float w01 = (1.f - ly) * lx         * my0 * mx1;
        const float w10 = ly         * (1.f - lx) * my1 * mx0;
        const float w11 = ly         * lx         * my1 * mx1;
        const int yc0 = y0 < 0 ? 0 : (y0 > Hn - 1 ? Hn - 1 : y0);
        const int yc1 = y1 < 0 ? 0 : (y1 > Hn - 1 ? Hn - 1 : y1);
        const int xc0 = x0 < 0 ? 0 : (x0 > Wn - 1 ? Wn - 1 : x0);
        const int xc1 = x1 < 0 ? 0 : (x1 > Wn - 1 ? Wn - 1 : x1);
        const float* p00 = xtb + ((size_t)yc0 * Wn + xc0) * Cn;
        const float* p01 = xtb + ((size_t)yc0 * Wn + xc1) * Cn;
        const float* p10 = xtb + ((size_t)yc1 * Wn + xc0) * Cn;
        const float* p11 = xtb + ((size_t)yc1 * Wn + xc1) * Cn;

#pragma unroll
        for (int half = 0; half < 2; ++half) {
            // B 32x16, contiguous k order chosen here (same as A): lanes 0-15 carry K 0..15, lanes 16-31 carry K 16..31
            const int cbase = half * 32 + laneHi * 16;
            float s[16];
#pragma unroll
            for (int q = 0; q < 4; ++q) {
                float4 a0 = *(const float4*)(p00 + cbase + 4 * q);
                float4 a1 = *(const float4*)(p01 + cbase + 4 * q);
                float4 a2 = *(const float4*)(p10 + cbase + 4 * q);
                float4 a3 = *(const float4*)(p11 + cbase + 4 * q);
                s[4 * q + 0] = w00 * a0.x + w01 * a1.x + w10 * a2.x + w11 * a3.x;
                s[4 * q + 1] = w00 * a0.y + w01 * a1.y + w10 * a2.y + w11 * a3.y;
                s[4 * q + 2] = w00 * a0.z + w01 * a1.z + w10 * a2.z + w11 * a3.z;
                s[4 * q + 3] = w00 * a0.w + w01 * a1.w + w10 * a2.w + w11 * a3.w;
            }
            v16h bf;
#pragma unroll
            for (int j = 0; j < 16; ++j) bf[j] = (_Float16)s[j];

            const v16h* Ac = (const v16h*)aLds + (size_t)(k2 * 2 + half) * 128;
#pragma unroll
            for (int mt = 0; mt < 4; ++mt) {
                v16h af = Ac[mt * 32 + lane];   // ds_load_b128 x2
                acc[mt] = __builtin_amdgcn_wmma_f32_16x16x32_f16(
                    false, af, false, bf, (short)0, acc[mt], false, false);
            }
        }
    }

    // D 16x16 f32 layout: VGPR r -> M = r (+8 for lanes 16-31), lane&15 -> N
    const size_t obase = (((size_t)b * COUT + laneHi * 8) * Hn + ho) * Wn + wo;
#pragma unroll
    for (int mt = 0; mt < 4; ++mt) {
#pragma unroll
        for (int r = 0; r < 8; ++r) {
            __builtin_nontemporal_store(acc[mt][r],
                                        &out[obase + (size_t)(mt * 16 + r) * HW]);
        }
    }
}

extern "C" void kernel_launch(void* const* d_in, const int* in_sizes, int n_in,
                              void* d_out, int out_size, void* d_ws, size_t ws_size,
                              hipStream_t stream) {
    const float* x      = (const float*)d_in[0];
    const float* offset = (const float*)d_in[1];
    const float* weight = (const float*)d_in[2];
    float* out = (float*)d_out;

    _Float16* aw = (_Float16*)d_ws;
    float*    xt = (float*)((char*)d_ws + AW_BYTES);

    stage_weights_kernel<<<9, 256, 0, stream>>>(weight, aw);
    nchw_to_nhwc_kernel<<<Bn * Hn, 256, 0, stream>>>(x, xt);
    deform_conv_wmma_kernel<<<NTILES / 8, 256, AW_BYTES, stream>>>(xt, offset, aw, out);
}
